// MultiheadAttention_44427141710474
// MI455X (gfx1250) — matrix operand order repaired and checked by reading, not compiled, not run on hardware
//
#include <hip/hip_runtime.h>
#include <hip/hip_bf16.h>

// ---------------------------------------------------------------------------
// MI455X (gfx1250) multi-head attention, WMMA f16 -> f32 accumulate.
// B=8, S=1024, D=1024, H=8, dh=128.
// ---------------------------------------------------------------------------

typedef __attribute__((ext_vector_type(16))) _Float16 v16h;
typedef __attribute__((ext_vector_type(8)))  _Float16 v8h;
typedef __attribute__((ext_vector_type(8)))  float    v8f;
typedef __attribute__((ext_vector_type(4)))  float    v4f;

#define WMMA_F16(a, b, c) \
  __builtin_amdgcn_wmma_f32_16x16x32_f16(false, (a), false, (b), (short)0, (c), false, false)

// A operand (16x32 f16): lane m, elements 0..7 = K[8*half .. +7],
//                        elements 8..15 = K[16+8*half .. +7].
// B operand (32x16 f16): lane n, elements 0..7 = K[8*half .. +7], 8..15 = K[16+8*half .. +7].

__device__ static inline v16h cat16(v8h a, v8h b) {
  v16h r;
#pragma unroll
  for (int i = 0; i < 8; ++i) { r[i] = a[i]; r[i + 8] = b[i]; }
  return r;
}

// ---------------------------------------------------------------------------
// Kernel 0: one-time fp32 -> f16 conversion (bandwidth pass).
// ---------------------------------------------------------------------------
__global__ __launch_bounds__(256) void cvt_kernel(const float* __restrict__ src,
                                                  _Float16* __restrict__ dst, int n) {
  const int i = (blockIdx.x * 256 + threadIdx.x) * 8;
  if (i < n) {
    v4f x0 = *(const v4f*)(src + i);
    v4f x1 = *(const v4f*)(src + i + 4);
    v8h r;
#pragma unroll
    for (int j = 0; j < 4; ++j) {
      r[j]     = (_Float16)x0[j];
      r[j + 4] = (_Float16)x1[j];
    }
    *(v8h*)(dst + i) = r;
  }
}

// ---------------------------------------------------------------------------
// Kernel 1: input projections, pure-f16 operands.
// out[m,n] = sum_k X[m,k]*W[n,k] + bias[n];  M=8192, N=1024, K=1024.
// Each wave: 32 rows x 64 cols (2 A-tiles x 4 B-tiles, 8 WMMAs/K-step).
// V output is stored transposed [B,H,dh,S] for later WMMA B operands.
// ---------------------------------------------------------------------------
__global__ __launch_bounds__(128) void proj_kernel(
    const _Float16* __restrict__ qh, const _Float16* __restrict__ kh,
    const _Float16* __restrict__ vh,
    const _Float16* __restrict__ Wqh, const float* __restrict__ bq,
    const _Float16* __restrict__ Wkh, const float* __restrict__ bk,
    const _Float16* __restrict__ Wvh, const float* __restrict__ bv,
    _Float16* __restrict__ Qp, _Float16* __restrict__ Kp, _Float16* __restrict__ VpT) {
  const int D = 1024, S = 1024, H = 8, dh = 128;
  const int lane = threadIdx.x & 31;
  const int widx = blockIdx.x * (blockDim.x >> 5) + (threadIdx.x >> 5);
  const int which = widx / 4096;   // 0=q 1=k 2=v
  const int t = widx % 4096;
  const int nt = t & 15;           // 16 groups of 64 columns
  const int mt = t >> 4;           // 256 groups of 32 rows
  const int m0 = mt * 32, n0 = nt * 64;
  const int ln = lane & 15, half = lane >> 4;

  const _Float16* X;
  const _Float16* W;
  const float* bias;
  if (which == 0)      { X = qh; W = Wqh; bias = bq; }
  else if (which == 1) { X = kh; W = Wkh; bias = bk; }
  else                 { X = vh; W = Wvh; bias = bv; }

  const _Float16* arow0 = X + (size_t)(m0 + ln) * D + 8 * half;
  const _Float16* arow1 = arow0 + (size_t)16 * D;
  const _Float16* brow  = W + (size_t)(n0 + ln) * D + 8 * half;

  v8f acc[2][4];
#pragma unroll
  for (int mi = 0; mi < 2; ++mi)
#pragma unroll
    for (int j = 0; j < 4; ++j) acc[mi][j] = (v8f){};

  for (int kc = 0; kc < D; kc += 32) {
    v8h a0l = *(const v8h*)(arow0 + kc);
    v8h a0h = *(const v8h*)(arow0 + kc + 16);
    v8h a1l = *(const v8h*)(arow1 + kc);
    v8h a1h = *(const v8h*)(arow1 + kc + 16);
    v8h br[4][2];
#pragma unroll
    for (int j = 0; j < 4; ++j) {
      const _Float16* p = brow + (size_t)(j * 16) * D + kc;
      br[j][0] = *(const v8h*)(p);
      br[j][1] = *(const v8h*)(p + 16);
    }
    const v16h A0 = cat16(a0l, a0h);
    const v16h A1 = cat16(a1l, a1h);
#pragma unroll
    for (int j = 0; j < 4; ++j) {
      const v16h Bt = cat16(br[j][0], br[j][1]);
      acc[0][j] = WMMA_F16(A0, Bt, acc[0][j]);
      acc[1][j] = WMMA_F16(A1, Bt, acc[1][j]);
    }
  }

#pragma unroll
  for (int mi = 0; mi < 2; ++mi) {
    const int mb = m0 + mi * 16;
#pragma unroll
    for (int j = 0; j < 4; ++j) {
      const int n = n0 + j * 16 + ln;
      const float bval = bias[n];
      if (which < 2) {
        _Float16* O = (which == 0) ? Qp : Kp;
#pragma unroll
        for (int r = 0; r < 8; ++r) {
          const int row = mb + 8 * half + r;
          O[(size_t)row * D + n] = (_Float16)(acc[mi][j][r] + bval);
        }
      } else {
        const int b_ = mb >> 10;
        const int s0 = (mb & 1023) + 8 * half;
        const int h_ = n >> 7;
        const int jj = n & 127;
        v8h pk;
#pragma unroll
        for (int r = 0; r < 8; ++r) pk[r] = (_Float16)(acc[mi][j][r] + bval);
        *(v8h*)(VpT + ((size_t)(b_ * H + h_) * dh + jj) * S + s0) = pk;
      }
    }
  }
}

// ---------------------------------------------------------------------------
// Kernel 2: one wave per (b, h, 16-query strip); 4 waves / 128-thread block.
// Phase 1: scores via WMMA, exact fp32 (incl -1e9 mask) written to raw out,
//          per-row max tracked in registers, broadcast via tiny LDS.
// Phase 2: re-read own raw strip (L2-hot) in A-layout, exp on the fly,
//          ctx = attn @ V via WMMA; row sums accumulated inline.
// Phase 3: normalize, 4KB LDS re-layout, out-proj ctx @ Wo^T + bo via WMMA.
// ---------------------------------------------------------------------------
__global__ __launch_bounds__(128) void attn_kernel(
    const _Float16* __restrict__ Qp, const _Float16* __restrict__ Kp,
    const _Float16* __restrict__ VpT, const int* __restrict__ mask,
    const _Float16* __restrict__ Woh, const float* __restrict__ bo,
    float* __restrict__ out, float* __restrict__ raw) {
  const int S = 1024, D = 1024, dh = 128, H = 8;
  __shared__ float    rowmaxs[4][16];
  __shared__ float    rowsums[4][16];
  __shared__ _Float16 ctxbuf[4][16 * 128];   // 4 KB per wave

  const int lane = threadIdx.x & 31;
  const int w = threadIdx.x >> 5;
  const int strip = blockIdx.x * 4 + w;      // 0..4095
  const int qt = strip & 63;
  const int h = (strip >> 6) & 7;
  const int b = strip >> 9;
  const int q0 = qt * 16;
  const int ln = lane & 15, half = lane >> 4;
  const float scale = 0.08838834764831845f;  // 1/sqrt(128)

  // Preload Q A-tiles (16 rows x 128 K) once.
  const _Float16* qrow = Qp + ((size_t)(b * S + q0 + ln)) * D + h * dh;
  v16h qA[4];
#pragma unroll
  for (int jc = 0; jc < 4; ++jc)
    qA[jc] = cat16(*(const v8h*)(qrow + jc * 32 + 8 * half),
                   *(const v8h*)(qrow + jc * 32 + 16 + 8 * half));

  float vmax[8];
#pragma unroll
  for (int r = 0; r < 8; ++r) vmax[r] = -3.4e38f;

  // ---- Phase 1: scores ----
  float* rawbase = raw + (size_t)(b * S + q0) * 8192 + (size_t)h * S;
  for (int kt = 0; kt < 64; ++kt) {
    const int k0 = kt * 16;
    const _Float16* krow = Kp + ((size_t)(b * S + k0 + ln)) * D + h * dh + 8 * half;
    v8h kb[4][2];
#pragma unroll
    for (int jc = 0; jc < 4; ++jc) {
      kb[jc][0] = *(const v8h*)(krow + jc * 32);
      kb[jc][1] = *(const v8h*)(krow + jc * 32 + 16);
    }
    v8f c = (v8f){};
#pragma unroll
    for (int jc = 0; jc < 4; ++jc) c = WMMA_F16(qA[jc], cat16(kb[jc][0], kb[jc][1]), c);

    const int kcol = k0 + ln;
    const int mv = mask[b * S + kcol];
    float* rp = rawbase + kcol + (size_t)(8 * half) * 8192;
#pragma unroll
    for (int r = 0; r < 8; ++r) {
      float sv = c[r] * scale;
      sv = (mv == 0) ? -1e9f : sv;
      rp[(size_t)r * 8192] = sv;                  // exact pre-softmax output
      vmax[r] = fmaxf(vmax[r], fmaxf(sv, -1.0e4f));
    }
  }

  // row-max reduce across 16 lanes of each half, publish to LDS
#pragma unroll
  for (int r = 0; r < 8; ++r) {
    float m = vmax[r];
    m = fmaxf(m, __shfl_xor(m, 1));
    m = fmaxf(m, __shfl_xor(m, 2));
    m = fmaxf(m, __shfl_xor(m, 4));
    m = fmaxf(m, __shfl_xor(m, 8));
    vmax[r] = m;
  }
  if (ln == 0) {
#pragma unroll
    for (int r = 0; r < 8; ++r) rowmaxs[w][r + 8 * half] = vmax[r];
  }
  const float mrow = rowmaxs[w][ln];   // max for this lane's A-layout row

  // ---- Phase 2: ctx = attn @ V, exp built from own raw strip ----
  v8f acc[8];
#pragma unroll
  for (int t = 0; t < 8; ++t) acc[t] = (v8f){};
  const float* rrow = raw + (size_t)(b * S + q0 + ln) * 8192 + (size_t)h * S;
  const _Float16* vbase = VpT + (size_t)(b * H + h) * dh * S + (size_t)ln * S + 8 * half;
  float lsum = 0.0f;

  for (int k0 = 0; k0 < S; k0 += 32) {
    v4f r0 = *(const v4f*)(rrow + k0 + 8 * half);
    v4f r1 = *(const v4f*)(rrow + k0 + 8 * half + 4);
    v4f r2 = *(const v4f*)(rrow + k0 + 16 + 8 * half);
    v4f r3 = *(const v4f*)(rrow + k0 + 16 + 8 * half + 4);
    v16h a;
#pragma unroll
    for (int i = 0; i < 4; ++i) {
      float e0 = __expf(fmaxf(r0[i], -1.0e4f) - mrow);
      float e1 = __expf(fmaxf(r1[i], -1.0e4f) - mrow);
      float e2 = __expf(fmaxf(r2[i], -1.0e4f) - mrow);
      float e3 = __expf(fmaxf(r3[i], -1.0e4f) - mrow);
      a[i] = (_Float16)e0; a[i + 4] = (_Float16)e1;
      a[i + 8] = (_Float16)e2; a[i + 12] = (_Float16)e3;
      lsum += e0 + e1 + e2 + e3;
    }
    // two groups of 4 V-tiles, loads batched ahead of the WMMA chain
#pragma unroll
    for (int g = 0; g < 2; ++g) {
      v8h vr[4][2];
#pragma unroll
      for (int t = 0; t < 4; ++t) {
        const _Float16* vp = vbase + (size_t)((g * 4 + t) * 16) * S + k0;
        vr[t][0] = *(const v8h*)(vp);
        vr[t][1] = *(const v8h*)(vp + 16);
      }
#pragma unroll
      for (int t = 0; t < 4; ++t)
        acc[g * 4 + t] = WMMA_F16(a, cat16(vr[t][0], vr[t][1]), acc[g * 4 + t]);
    }
  }

  // row sums: lane l and l+16 cover complementary chunks of row l&15
  lsum += __shfl_xor(lsum, 16);
  if (half == 0) rowsums[w][ln] = lsum;
  float inv[8];
#pragma unroll
  for (int r = 0; r < 8; ++r) inv[r] = 1.0f / rowsums[w][r + 8 * half];

  // ---- Phase 3: normalize, re-layout through LDS, output projection ----
  _Float16* cb = ctxbuf[w];
#pragma unroll
  for (int t = 0; t < 8; ++t) {
#pragma unroll
    for (int r = 0; r < 8; ++r)
      cb[(r + 8 * half) * 128 + t * 16 + ln] = (_Float16)(acc[t][r] * inv[r]);
  }

#pragma unroll
  for (int nt = 0; nt < 8; ++nt) {
    const int e = nt * 16 + ln;
    const float bval = bo[e];
    v8f c;
#pragma unroll
    for (int r = 0; r < 8; ++r) c[r] = bval;
#pragma unroll
    for (int kc = 0; kc < 4; ++kc) {
      const _Float16* wrow = Woh + (size_t)e * 128 + kc * 32 + 8 * half;
      v16h btile = cat16(*(const v8h*)(wrow), *(const v8h*)(wrow + 16));
      v16h atile = cat16(*(const v8h*)(cb + ln * 128 + kc * 32 + 8 * half),
                         *(const v8h*)(cb + ln * 128 + kc * 32 + 16 + 8 * half));
      c = WMMA_F16(atile, btile, c);
    }
    float* op = out + (size_t)(b * S + q0 + 8 * half) * D + h * 128 + e;
#pragma unroll
    for (int r = 0; r < 8; ++r) op[(size_t)r * D] = c[r];
  }
}

// ---------------------------------------------------------------------------
extern "C" void kernel_launch(void* const* d_in, const int* in_sizes, int n_in,
                              void* d_out, int out_size, void* d_ws, size_t ws_size,
                              hipStream_t stream) {
  (void)in_sizes; (void)n_in; (void)out_size; (void)ws_size;
  const float* q    = (const float*)d_in[0];
  const float* k    = (const float*)d_in[1];
  const float* v    = (const float*)d_in[2];
  const int*   mask = (const int*)d_in[3];
  const float* Wq = (const float*)d_in[4];
  const float* bq = (const float*)d_in[5];
  const float* Wk = (const float*)d_in[6];
  const float* bk = (const float*)d_in[7];
  const float* Wv = (const float*)d_in[8];
  const float* bv = (const float*)d_in[9];
  const float* Wo = (const float*)d_in[10];
  const float* bo = (const float*)d_in[11];

  float* out = (float*)d_out;                       // [8,1024,1024]
  float* raw = out + (size_t)8 * 1024 * 1024;       // [8,1024,8192]

  const size_t NQ = (size_t)8 * 1024 * 1024;        // 8M elements
  const size_t NW = (size_t)1024 * 1024;            // 1M elements
  _Float16* Qp  = (_Float16*)d_ws;                  // projected Q (f16)
  _Float16* Kp  = Qp + NQ;                          // projected K (f16)
  _Float16* VpT = Kp + NQ;                          // projected V, transposed
  _Float16* qh  = VpT + NQ;                         // f16 copies of inputs
  _Float16* kh  = qh + NQ;
  _Float16* vh  = kh + NQ;
  _Float16* Wqh = vh + NQ;                          // f16 copies of weights
  _Float16* Wkh = Wqh + NW;
  _Float16* Wvh = Wkh + NW;
  _Float16* Woh = Wvh + NW;                         // 128x128

  // one-time fp32 -> f16 conversions
  cvt_kernel<<<4096, 256, 0, stream>>>(q, qh, (int)NQ);
  cvt_kernel<<<4096, 256, 0, stream>>>(k, kh, (int)NQ);
  cvt_kernel<<<4096, 256, 0, stream>>>(v, vh, (int)NQ);
  cvt_kernel<<<512, 256, 0, stream>>>(Wq, Wqh, (int)NW);
  cvt_kernel<<<512, 256, 0, stream>>>(Wk, Wkh, (int)NW);
  cvt_kernel<<<512, 256, 0, stream>>>(Wv, Wvh, (int)NW);
  cvt_kernel<<<8, 256, 0, stream>>>(Wo, Woh, 128 * 128);

  // 3 projections * 4096 waves each, 4 waves per 128-thread block
  proj_kernel<<<3072, 128, 0, stream>>>(qh, kh, vh, Wqh, bq, Wkh, bk, Wvh, bv, Qp, Kp, VpT);
  // one wave per (b, h, 16-query strip)
  attn_kernel<<<1024, 128, 0, stream>>>(Qp, Kp, VpT, mask, Woh, bo, out, raw);
}
